// RNNStackJoulin_39951785788096
// MI455X (gfx1250) — matrix operand order repaired and checked by reading, not compiled, not run on hardware
//
#include <hip/hip_runtime.h>

typedef __attribute__((ext_vector_type(16))) __bf16 v16bf;
typedef __attribute__((ext_vector_type(8)))  __bf16 v8bf;
typedef __attribute__((ext_vector_type(8)))  float  v8f;

#define B_  32
#define S_  512
#define V_  4096
#define E_  128
#define H_  256
#define KP  3
#define NS  20

// ---------- WMMA fragment helpers (CDNA5 16-bit layouts, wave32) ----------
// A 16x32 bf16: lane m (lane&15), hi = lane>>4.
//   elem j -> K = kb + 16*hi + j (32 contiguous bytes, matches B)
__device__ __forceinline__ v16bf load_a_frag(const __bf16* A, int lda, int row0, int kb) {
  int lane = threadIdx.x & 31;
  int m = lane & 15, hi = lane >> 4;
  const __bf16* p = A + (long)(row0 + m) * lda + kb + 16 * hi;
  v8bf lo = *(const v8bf*)p;
  v8bf hh = *(const v8bf*)(p + 8);
  return __builtin_shufflevector(lo, hh, 0,1,2,3,4,5,6,7,8,9,10,11,12,13,14,15);
}

// B 32x16 bf16 from transposed weights BT[N][K]: lane n holds K = kb + 16*hi + j
// -> 32 contiguous bytes per lane.
__device__ __forceinline__ v16bf load_b_frag(const __bf16* BT, int ldb, int col0, int kb) {
  int lane = threadIdx.x & 31;
  int n = lane & 15, hi = lane >> 4;
  const __bf16* p = BT + (long)(col0 + n) * ldb + kb + 16 * hi;
  v8bf lo = *(const v8bf*)p;
  v8bf hh = *(const v8bf*)(p + 8);
  return __builtin_shufflevector(lo, hh, 0,1,2,3,4,5,6,7,8,9,10,11,12,13,14,15);
}

__device__ __forceinline__ v8f wmma_bf16(v16bf a, v16bf b, v8f c) {
  return __builtin_amdgcn_wmma_f32_16x16x32_bf16(false, a, false, b, (short)0, c, false, false);
}

__device__ __forceinline__ float sigmoidf_(float x) {
  return 1.f / (1.f + __expf(-x));
}

// ---------- kernel 0: f32 [K][N] weights -> bf16 transposed [N][K] ----------
__global__ void transpose_to_bf16(const float* __restrict__ src, __bf16* __restrict__ dst,
                                  int K, int N) {
  long idx = (long)blockIdx.x * blockDim.x + threadIdx.x;
  if (idx >= (long)K * N) return;
  int nn = (int)(idx / K), kk = (int)(idx % K);
  dst[idx] = (__bf16)src[(long)kk * N + nn];
}

// ---------- kernel 1: gather emb rows, L2-normalize, store bf16 time-major ----------
__global__ void embed_norm_kernel(const int* __restrict__ x_ids, const float* __restrict__ emb,
                                  __bf16* __restrict__ xn) {
  int wave = threadIdx.x >> 5, lane = threadIdx.x & 31;
  int r = blockIdx.x * 8 + wave;          // r = s*B + b
  int s = r >> 5, b = r & 31;
  int id = x_ids[b * S_ + s];
  const float* e = emb + (long)id * E_ + lane * 4;
  float4 v = *(const float4*)e;
  float ss = v.x * v.x + v.y * v.y + v.z * v.z + v.w * v.w;
  #pragma unroll
  for (int m = 16; m >= 1; m >>= 1) ss += __shfl_xor(ss, m, 32);
  float sc = 1.f / fmaxf(sqrtf(ss), 1e-12f);
  __bf16* o = xn + (long)r * E_ + lane * 4;
  o[0] = (__bf16)(v.x * sc); o[1] = (__bf16)(v.y * sc);
  o[2] = (__bf16)(v.z * sc); o[3] = (__bf16)(v.w * sc);
}

// ---------- kernel 2: ri = xn @ W_i  (M=16384,N=256,K=128), f32 out ----------
__global__ void ri_gemm_kernel(const __bf16* __restrict__ xn, const __bf16* __restrict__ WiT,
                               float* __restrict__ ri) {
  int wave = threadIdx.x >> 5;
  int wg = blockIdx.x * 8 + wave;          // 4096 waves total
  int mt = wg >> 2, cg = wg & 3;
  int row0 = mt * 16, col0 = cg * 64;
  v8f acc[4] = {};
  for (int kb = 0; kb < E_; kb += 32) {
    v16bf a = load_a_frag(xn, E_, row0, kb);
    #pragma unroll
    for (int q = 0; q < 4; ++q) {
      v16bf bb = load_b_frag(WiT, E_, col0 + q * 16, kb);
      acc[q] = wmma_bf16(a, bb, acc[q]);
    }
  }
  int lane = threadIdx.x & 31, n = lane & 15, hi = lane >> 4;
  #pragma unroll
  for (int q = 0; q < 4; ++q)
    #pragma unroll
    for (int r = 0; r < 8; ++r)
      ri[(long)(row0 + 8 * hi + r) * H_ + col0 + q * 16 + n] = acc[q][r];
}

// ---------- kernel 3: sequential recurrence, single 8-wave workgroup ----------
__global__ void __launch_bounds__(256) recurrent_kernel(
    const float* __restrict__ ri, const __bf16* __restrict__ WhT,
    const __bf16* __restrict__ WsT, const float* __restrict__ W_push,
    const float* __restrict__ W_ops, __bf16* __restrict__ h_all,
    __bf16* stackA, __bf16* stackB) {
  __shared__ float  hf[B_ * H_];          // 32 KB
  __shared__ __bf16 hb[B_ * H_];          // 16 KB
  __shared__ __bf16 peek[B_ * KP * E_];   // 24 KB
  __shared__ float  pushv[B_ * E_];       // 16 KB
  __shared__ float  opsl[B_ * 3];
  __shared__ float  opsv[B_ * 3];

  int t = threadIdx.x, lane = t & 31, wave = t >> 5;
  int n = lane & 15, hi = lane >> 4;

  for (int i = t; i < B_ * H_; i += 256) { hf[i] = 0.f; hb[i] = (__bf16)0.f; }
  for (int i = t; i < B_ * NS * E_; i += 256) stackA[i] = (__bf16)0.f;
  __syncthreads();

  for (int s = 0; s < S_; ++s) {
    __bf16* cur = (s & 1) ? stackB : stackA;
    __bf16* nxt = (s & 1) ? stackA : stackB;

    // peek = stack[:, :3, :] -> LDS (contiguous 384 per batch row)
    for (int i = t; i < B_ * KP * E_; i += 256) {
      int b = i / (KP * E_), j = i % (KP * E_);
      peek[i] = cur[b * NS * E_ + j];
    }
    __syncthreads();

    // pre = ri[s] + h @ W_h + peek @ W_s ; 32 tiles of 16x16, 4 per wave
    v8f acc[4];
    #pragma unroll
    for (int q = 0; q < 4; ++q) {
      int tile = wave * 4 + q;
      int mt = tile >> 4, nt = tile & 15;
      int row0 = mt * 16, col0 = nt * 16;
      v8f c;
      #pragma unroll
      for (int r = 0; r < 8; ++r)
        c[r] = ri[((long)s * B_ + row0 + 8 * hi + r) * H_ + col0 + n];
      for (int kb = 0; kb < H_; kb += 32) {
        v16bf a  = load_a_frag(hb, H_, row0, kb);
        v16bf bb = load_b_frag(WhT, H_, col0, kb);
        c = wmma_bf16(a, bb, c);
      }
      for (int kb = 0; kb < KP * E_; kb += 32) {
        v16bf a  = load_a_frag(peek, KP * E_, row0, kb);
        v16bf bb = load_b_frag(WsT, KP * E_, col0, kb);
        c = wmma_bf16(a, bb, c);
      }
      acc[q] = c;
    }
    __syncthreads();   // all WMMA reads of hb complete before overwrite

    #pragma unroll
    for (int q = 0; q < 4; ++q) {
      int tile = wave * 4 + q;
      int mt = tile >> 4, nt = tile & 15;
      int row0 = mt * 16, col0 = nt * 16;
      #pragma unroll
      for (int r = 0; r < 8; ++r) {
        int row = row0 + 8 * hi + r, col = col0 + n;
        float hval = sigmoidf_(acc[q][r]);
        hf[row * H_ + col] = hval;
        __bf16 hv = (__bf16)hval;
        hb[row * H_ + col] = hv;
        h_all[((long)s * B_ + row) * H_ + col] = hv;
      }
    }
    __syncthreads();

    // ops logits (32x3) — tiny, scalar VALU
    if (t < B_ * 3) {
      int b = t / 3, j = t % 3;
      float a2 = 0.f;
      for (int k = 0; k < H_; ++k) a2 += hf[b * H_ + k] * W_ops[k * 3 + j];
      opsl[t] = a2;
    }
    // push_val = sigmoid(h @ W_push) (32x128)
    for (int i = t; i < B_ * E_; i += 256) {
      int b = i / E_, e = i % E_;
      float a2 = 0.f;
      for (int k = 0; k < H_; ++k) a2 += hf[b * H_ + k] * W_push[k * E_ + e];
      pushv[i] = sigmoidf_(a2);
    }
    __syncthreads();
    if (t < B_) {
      float a0 = opsl[t * 3], a1 = opsl[t * 3 + 1], a2 = opsl[t * 3 + 2];
      float mm = fmaxf(a0, fmaxf(a1, a2));
      float e0 = __expf(a0 - mm), e1 = __expf(a1 - mm), e2 = __expf(a2 - mm);
      float inv = 1.f / (e0 + e1 + e2);
      opsv[t * 3] = e0 * inv; opsv[t * 3 + 1] = e1 * inv; opsv[t * 3 + 2] = e2 * inv;
    }
    __syncthreads();

    // differentiable stack blend (ping-pong in global scratch, L2-resident)
    for (int i = t; i < B_ * NS * E_; i += 256) {
      int b = i / (NS * E_);
      int rest = i % (NS * E_);
      int slot = rest / E_, e = rest % E_;
      float push = opsv[b * 3], pop = opsv[b * 3 + 1], nop = opsv[b * 3 + 2];
      float pushed = (slot == 0) ? pushv[b * E_ + e]
                                 : (float)cur[b * NS * E_ + (slot - 1) * E_ + e];
      float popped = (slot < NS - 1) ? (float)cur[b * NS * E_ + (slot + 1) * E_ + e] : 0.f;
      float same = (float)cur[i];
      nxt[i] = (__bf16)(push * pushed + pop * popped + nop * same);
    }
    __syncthreads();
  }
}

// ---------- kernel 4: logits = h_all @ W_fc (M=16384,N=4096,K=256) -> d_out ----------
// Block = 128 rows x 64 cols. The 64x256 bf16 B-panel (32 KB) is staged into LDS
// once per block with GLOBAL_LOAD_ASYNC_TO_LDS_B128 (ASYNCcnt) and reused by all
// 8 waves; inner loop reads B via ds_load_b128 and A via global_load_b128.
__global__ void __launch_bounds__(256) fc_gemm_kernel(
    const __bf16* __restrict__ h_all, const __bf16* __restrict__ WfcT,
    float* __restrict__ out) {
  __shared__ __align__(16) __bf16 Bs[64 * H_];   // 32 KB panel [64 N-rows][256 K]

  int t = threadIdx.x, wave = t >> 5;
  int rowBlock = blockIdx.x >> 6, colBlock = blockIdx.x & 63;
  int row0 = rowBlock * 128 + wave * 16;
  int col0 = colBlock * 64;

  // --- async stage B panel: 64 rows x 512B = 2048 x 16B chunks, 8 per thread ---
  {
    unsigned lds_base = (unsigned)(size_t)(void*)Bs;      // low 32 bits = LDS offset
    const char* gbase = (const char*)(WfcT + (long)col0 * H_);
    #pragma unroll
    for (int it = 0; it < 8; ++it) {
      int c = t + it * 256;
      unsigned la = lds_base + (unsigned)c * 16u;
      const char* ga = gbase + (long)c * 16;
      asm volatile("global_load_async_to_lds_b128 %0, %1, off"
                   :
                   : "v"(la), "v"(ga)
                   : "memory");
    }
    asm volatile("s_wait_asynccnt 0x0" ::: "memory");
  }
  __syncthreads();

  // --- 16x64 output tile per wave, A double-buffered across k-steps ---
  v8f acc[4] = {};
  v16bf a_cur = load_a_frag(h_all, H_, row0, 0);
  for (int kb = 0; kb < H_; kb += 32) {
    v16bf a_nxt;
    if (kb + 32 < H_) a_nxt = load_a_frag(h_all, H_, row0, kb + 32);
    #pragma unroll
    for (int q = 0; q < 4; ++q) {
      v16bf bb = load_b_frag(Bs, H_, q * 16, kb);
      acc[q] = wmma_bf16(a_cur, bb, acc[q]);
    }
    a_cur = a_nxt;
  }

  int lane = t & 31, n = lane & 15, hi = lane >> 4;
  #pragma unroll
  for (int q = 0; q < 4; ++q)
    #pragma unroll
    for (int r = 0; r < 8; ++r) {
      int row = row0 + 8 * hi + r;         // row = s*B + b (time-major)
      int s = row >> 5, b = row & 31;
      out[((long)b * S_ + s) * V_ + col0 + q * 16 + n] = acc[q][r];
    }
}

// ---------- kernel 5: in-place row softmax over d_out (rows of 4096) ----------
__global__ void softmax_kernel(float* __restrict__ out) {
  __shared__ float red[256];
  int t = threadIdx.x;
  float* p = out + (long)blockIdx.x * V_;
  float m = -3.4e38f;
  for (int i = t; i < V_; i += 256) m = fmaxf(m, p[i]);
  red[t] = m; __syncthreads();
  for (int st = 128; st; st >>= 1) { if (t < st) red[t] = fmaxf(red[t], red[t + st]); __syncthreads(); }
  m = red[0]; __syncthreads();
  float sum = 0.f;
  for (int i = t; i < V_; i += 256) sum += __expf(p[i] - m);
  red[t] = sum; __syncthreads();
  for (int st = 128; st; st >>= 1) { if (t < st) red[t] += red[t + st]; __syncthreads(); }
  float inv = 1.f / red[0];
  for (int i = t; i < V_; i += 256) p[i] = __expf(p[i] - m) * inv;
}

extern "C" void kernel_launch(void* const* d_in, const int* in_sizes, int n_in,
                              void* d_out, int out_size, void* d_ws, size_t ws_size,
                              hipStream_t stream) {
  const int*   x_ids  = (const int*)  d_in[0];
  const float* emb    = (const float*)d_in[1];
  const float* W_i    = (const float*)d_in[2];
  const float* W_h    = (const float*)d_in[3];
  const float* W_s    = (const float*)d_in[4];
  const float* W_fc   = (const float*)d_in[5];
  const float* W_push = (const float*)d_in[6];
  const float* W_ops  = (const float*)d_in[7];
  float* out = (float*)d_out;

  char* ws = (char*)d_ws;
  size_t off = 0;
  auto alloc = [&](size_t bytes) -> void* {
    void* p = ws + off;
    off = (off + bytes + 255) & ~(size_t)255;
    return p;
  };
  __bf16* xn     = (__bf16*)alloc((size_t)S_ * B_ * E_ * sizeof(__bf16));
  __bf16* WiT    = (__bf16*)alloc((size_t)H_ * E_ * sizeof(__bf16));
  __bf16* WhT    = (__bf16*)alloc((size_t)H_ * H_ * sizeof(__bf16));
  __bf16* WsT    = (__bf16*)alloc((size_t)H_ * KP * E_ * sizeof(__bf16));
  __bf16* WfcT   = (__bf16*)alloc((size_t)V_ * H_ * sizeof(__bf16));
  float*  ri     = (float*) alloc((size_t)S_ * B_ * H_ * sizeof(float));
  __bf16* h_all  = (__bf16*)alloc((size_t)S_ * B_ * H_ * sizeof(__bf16));
  __bf16* stackA = (__bf16*)alloc((size_t)B_ * NS * E_ * sizeof(__bf16));
  __bf16* stackB = (__bf16*)alloc((size_t)B_ * NS * E_ * sizeof(__bf16));
  (void)ws_size; (void)in_sizes; (void)n_in; (void)out_size;

  // weight convert + transpose to bf16 [N][K]
  transpose_to_bf16<<<(E_ * H_ + 255) / 256, 256, 0, stream>>>(W_i, WiT, E_, H_);
  transpose_to_bf16<<<(H_ * H_ + 255) / 256, 256, 0, stream>>>(W_h, WhT, H_, H_);
  transpose_to_bf16<<<(KP * E_ * H_ + 255) / 256, 256, 0, stream>>>(W_s, WsT, KP * E_, H_);
  transpose_to_bf16<<<(H_ * V_ + 255) / 256, 256, 0, stream>>>(W_fc, WfcT, H_, V_);

  // embedding gather + L2 normalize (one wave per row)
  embed_norm_kernel<<<(S_ * B_) / 8, 256, 0, stream>>>(x_ids, emb, xn);

  // ri = xn @ W_i
  ri_gemm_kernel<<<((S_ * B_ / 16) * 4) / 8, 256, 0, stream>>>(xn, WiT, ri);

  // sequential recurrence (persistent single workgroup)
  recurrent_kernel<<<1, 256, 0, stream>>>(ri, WhT, WsT, W_push, W_ops, h_all, stackA, stackB);

  // logits = h_all @ W_fc into d_out (LDS-staged B panel), then in-place softmax
  fc_gemm_kernel<<<(S_ * B_ / 128) * (V_ / 64), 256, 0, stream>>>(h_all, WfcT, out);
  softmax_kernel<<<S_ * B_, 256, 0, stream>>>(out);
}
